// TextAttentionBlock_51161650430486
// MI455X (gfx1250) — matrix operand order repaired and checked by reading, not compiled, not run on hardware
//
#include <hip/hip_runtime.h>
#include <hip/hip_bf16.h>

typedef __attribute__((ext_vector_type(16))) _Float16 v16h;
typedef __attribute__((ext_vector_type(8)))  _Float16 v8h;
typedef __attribute__((ext_vector_type(2)))  _Float16 v2h;
typedef __attribute__((ext_vector_type(8)))  float    v8f;

#define C_CH 512
#define HWTOK 4096
#define NHEAD 8
#define HDIM 64
#define LTXT 77
#define LPAD 96
#define TT 64       // tokens per workgroup tile
#define XS 528      // LDS half-stride for [token][channel] tiles (32B aligned, bank-staggered)
#define PSS 96      // LDS half-stride for P tiles

static __device__ __forceinline__ v16h cat16(v8h lo, v8h hi) {
    return __builtin_shufflevector(lo, hi, 0,1,2,3,4,5,6,7,8,9,10,11,12,13,14,15);
}

// A-fragment (16x32 f16, M rows of a row-major matrix): per-lane two contiguous 16B loads.
static __device__ __forceinline__ v16h loadA(const _Float16* rowp, int kc, int lane) {
    int base = (lane >> 4) << 4;                 // 0 or 16
    v8h lo = *(const v8h*)(rowp + kc + base);
    v8h hi = *(const v8h*)(rowp + kc + base + 8);
    return cat16(lo, hi);
}

// B-fragment (32x16 f16, K x N) from an N-major [n][k] array: one contiguous 32B load per lane.
static __device__ __forceinline__ v16h loadB(const _Float16* basep, int strideH, int kc, int lane) {
    int n = lane & 15;
    int koff = kc + ((lane >> 4) << 4);          // +0 or +16
    return *(const v16h*)(basep + (size_t)n * strideH + koff);
}

static __device__ __forceinline__ v8f wmma16(v16h a, v16h b, v8f c) {
    return __builtin_amdgcn_wmma_f32_16x16x32_f16(false, a, false, b, (short)0, c, false, false);
}

// ---------------------------------------------------------------- P1: f32 -> f16 weight convert
__global__ void cvt_f16_kernel(const float* __restrict__ src, _Float16* __restrict__ dst, int n) {
    int i = blockIdx.x * blockDim.x + threadIdx.x;
    if (i < n) dst[i] = (_Float16)src[i];
}

// ---------------------------------------------------------------- P2: K/V projections via WMMA
// out: kf32 [b][LPAD][512] (f32, rows >=77 untouched), vhT [b][head][hd][LPAD] (f16)
__global__ __launch_bounds__(256)
void kv_gemm_kernel(const float* __restrict__ text,
                    const _Float16* __restrict__ wk16, const _Float16* __restrict__ wv16,
                    const float* __restrict__ bk, const float* __restrict__ bv,
                    float* __restrict__ kf32, _Float16* __restrict__ vhT) {
    int tid = threadIdx.x, lane = tid & 31;
    int wid = blockIdx.x * 8 + (tid >> 5);   // 0..2559 = 16b * 32ct * 5lt
    int b = wid / 160;
    int rem = wid % 160;
    int ct = rem / 5, lt = rem % 5;
    int mrow = lane & 15;

    v8f Ck = {0.f,0.f,0.f,0.f,0.f,0.f,0.f,0.f};
    v8f Cv = Ck;

    int l = lt * 16 + (lane & 15);
    bool lvalid = (l < LTXT);
    const float* trow = text + ((size_t)b * LTXT + (lvalid ? l : 0)) * 768;
    const _Float16* akrow = wk16 + (size_t)(ct * 16 + mrow) * 768;
    const _Float16* avrow = wv16 + (size_t)(ct * 16 + mrow) * 768;

    for (int d = 0; d < 768; d += 32) {
        int koff = d + ((lane >> 4) << 4);
        v16h bt;
        if (lvalid) {
            #pragma unroll
            for (int i = 0; i < 16; ++i) bt[i] = (_Float16)trow[koff + i];
        } else {
            #pragma unroll
            for (int i = 0; i < 16; ++i) bt[i] = (_Float16)0.0f;
        }
        v16h ak = loadA(akrow, d, lane);
        v16h av = loadA(avrow, d, lane);
        Ck = wmma16(ak, bt, Ck);
        Cv = wmma16(av, bt, Cv);
    }

    if (lvalid) {
        int mhi = (lane >> 4) << 3;
        #pragma unroll
        for (int r = 0; r < 8; ++r) {
            int c = ct * 16 + r + mhi;
            kf32[((size_t)b * LPAD + l) * C_CH + c] = Ck[r] + bk[c];
            float vv = Cv[r] + bv[c];
            vhT[(((size_t)b * NHEAD + (c >> 6)) * HDIM + (c & 63)) * LPAD + l] = (_Float16)vv;
        }
    }
}

// ---------------------------------------------------------------- P3: LayerNorm(K) -> kh f16, zero pads
// kh [b][head][LPAD][hd]
__global__ __launch_bounds__(256)
void ln_k_kernel(const float* __restrict__ kf32, const float* __restrict__ g2,
                 const float* __restrict__ b2,
                 _Float16* __restrict__ kh, _Float16* __restrict__ vhT) {
    __shared__ float s1[256], s2[256];
    int b = blockIdx.x / LPAD;
    int l = blockIdx.x % LPAD;
    int tid = threadIdx.x;
    if (l >= LTXT) {
        for (int c = tid; c < C_CH; c += 256) {
            kh[(((size_t)b * NHEAD + (c >> 6)) * LPAD + l) * HDIM + (c & 63)] = (_Float16)0.0f;
            vhT[(((size_t)b * NHEAD + (c >> 6)) * HDIM + (c & 63)) * LPAD + l] = (_Float16)0.0f;
        }
        return;
    }
    const float* kp = kf32 + ((size_t)b * LPAD + l) * C_CH;
    float v0 = kp[tid], v1 = kp[tid + 256];
    s1[tid] = v0 + v1;
    s2[tid] = v0 * v0 + v1 * v1;
    __syncthreads();
    for (int off = 128; off > 0; off >>= 1) {
        if (tid < off) { s1[tid] += s1[tid + off]; s2[tid] += s2[tid + off]; }
        __syncthreads();
    }
    float mean = s1[0] * (1.0f / C_CH);
    float var = s2[0] * (1.0f / C_CH) - mean * mean;
    float rstd = rsqrtf(var + 1e-5f);
    #pragma unroll
    for (int it = 0; it < 2; ++it) {
        int c = tid + it * 256;
        float v = it ? v1 : v0;
        float o = (v - mean) * rstd * g2[c] + b2[c];
        kh[(((size_t)b * NHEAD + (c >> 6)) * LPAD + l) * HDIM + (c & 63)] = (_Float16)o;
    }
}

// ---------------------------------------------------------------- main fused kernel
// one workgroup per (batch, 64-token tile); 8 waves = 8 heads for attention phase
__global__ __launch_bounds__(256)
void fused_attn_kernel(const float* __restrict__ x,
                       const _Float16* __restrict__ wq16, const _Float16* __restrict__ wo16,
                       const _Float16* __restrict__ kh, const _Float16* __restrict__ vhT,
                       const float* __restrict__ bq, const float* __restrict__ bo,
                       const float* __restrict__ g1, const float* __restrict__ b1,
                       float* __restrict__ out) {
    __shared__ alignas(32) _Float16 xs_os[TT * XS];          // x tile (f16), later reused for attn out
    __shared__ alignas(32) _Float16 qh[TT * XS];             // q tokens post-LN (f16)
    __shared__ alignas(32) _Float16 ps[NHEAD][32 * PSS];     // per-head softmax P (f16), 32-token half
    __shared__ float red1[TT][4], red2[TT][4];

    const int tid = threadIdx.x;
    const int lane = tid & 31;
    const int wav = tid >> 5;                 // wave id / head id
    const int b = blockIdx.x >> 6;            // /64
    const int t0 = (blockIdx.x & 63) << 6;    // *64
    const int mrow = lane & 15;
    const int mhi = (lane >> 4) << 3;
    const int obase = wav * 64;

    // --- stage x tile -> xs_os[t][c] (f16); each wave owns a contiguous 64-channel slab ---
    {
        const float* xb = x + (size_t)b * C_CH * HWTOK + t0;
        #pragma unroll
        for (int th = 0; th < 2; ++th) {
            int t = lane + th * 32;
            for (int c0 = obase; c0 < obase + 64; c0 += 2) {
                float f0 = xb[(size_t)c0 * HWTOK + t];
                float f1 = xb[(size_t)(c0 + 1) * HWTOK + t];
                v2h h; h[0] = (_Float16)f0; h[1] = (_Float16)f1;
                *(v2h*)&xs_os[t * XS + c0] = h;
            }
        }
    }
    // zero P pad columns (wave-private buffer; stays zero across both halves)
    {
        uint32_t* pz = (uint32_t*)&ps[wav][lane * PSS + 80];
        #pragma unroll
        for (int i = 0; i < 8; ++i) pz[i] = 0u;
    }
    __syncthreads();

    // --- GEMM1: q[o][t] = wq * x ; wave covers 64 o-channels x 64 tokens ---
    v8f Cq[4][4];
    {
        v8f z = {0.f,0.f,0.f,0.f,0.f,0.f,0.f,0.f};
        #pragma unroll
        for (int i = 0; i < 4; ++i)
            #pragma unroll
            for (int j = 0; j < 4; ++j) Cq[i][j] = z;
    }
    for (int kc = 0; kc < C_CH; kc += 32) {
        v16h bfr[4];
        #pragma unroll
        for (int tt = 0; tt < 4; ++tt) bfr[tt] = loadB(xs_os + tt * 16 * XS, XS, kc, lane);
        #pragma unroll
        for (int ot = 0; ot < 4; ++ot) {
            v16h a = loadA(wq16 + (size_t)(obase + ot * 16 + mrow) * C_CH, kc, lane);
            #pragma unroll
            for (int tt = 0; tt < 4; ++tt) Cq[ot][tt] = wmma16(a, bfr[tt], Cq[ot][tt]);
        }
    }
    // epilogue: +bias +positional encoding, write qh[t][o] as packed v8h rows
    {
        int n = lane & 15;
        #pragma unroll
        for (int tt = 0; tt < 4; ++tt) {
            int tl = tt * 16 + n;
            int tg = t0 + tl;
            float pex = 0.05f * ((float)(tg & 63) * (1.0f / 63.0f));
            float pey = 0.05f * ((float)(tg >> 6) * (1.0f / 63.0f));
            #pragma unroll
            for (int ot = 0; ot < 4; ++ot) {
                int o0 = obase + ot * 16 + mhi;
                v8h hv;
                #pragma unroll
                for (int r = 0; r < 8; ++r) {
                    int o = o0 + r;
                    hv[r] = (_Float16)(Cq[ot][tt][r] + bq[o] + ((o < 256) ? pex : pey));
                }
                *(v8h*)&qh[tl * XS + o0] = hv;
            }
        }
    }
    __syncthreads();

    // --- LayerNorm over channels per token (4 threads per token, v8h chunks) ---
    {
        int tl = tid >> 2, p = tid & 3;
        int cb = p * 128;
        float s = 0.f, s2 = 0.f;
        for (int c0 = cb; c0 < cb + 128; c0 += 8) {
            v8h h = *(const v8h*)&qh[tl * XS + c0];
            #pragma unroll
            for (int i = 0; i < 8; ++i) { float v = (float)h[i]; s += v; s2 += v * v; }
        }
        red1[tl][p] = s; red2[tl][p] = s2;
        __syncthreads();
        float ts = 0.f, ts2 = 0.f;
        #pragma unroll
        for (int i = 0; i < 4; ++i) { ts += red1[tl][i]; ts2 += red2[tl][i]; }
        float mean = ts * (1.0f / C_CH);
        float var = ts2 * (1.0f / C_CH) - mean * mean;
        float rstd = rsqrtf(var + 1e-5f);
        for (int c0 = cb; c0 < cb + 128; c0 += 8) {
            v8h h = *(const v8h*)&qh[tl * XS + c0];
            v8h o;
            #pragma unroll
            for (int i = 0; i < 8; ++i) {
                int c = c0 + i;
                o[i] = (_Float16)(((float)h[i] - mean) * rstd * g1[c] + b1[c]);
            }
            *(v8h*)&qh[tl * XS + c0] = o;
        }
    }
    __syncthreads();

    // --- attention: wave = head; two 32-token half-passes (wave-private P buffer) ---
    {
        const _Float16* khb = kh + ((size_t)b * NHEAD + wav) * LPAD * HDIM;   // [l][hd]
        const _Float16* vb = vhT + ((size_t)b * NHEAD + wav) * HDIM * LPAD;   // [hd][l]
        const float scale = 0.125f;  // HD^-0.5
        int coln = lane & 15;
        #pragma unroll
        for (int half = 0; half < 2; ++half) {
            int th0 = half * 32;
            // S = q . k^T
            v8f S[2][5];
            {
                v8f z = {0.f,0.f,0.f,0.f,0.f,0.f,0.f,0.f};
                #pragma unroll
                for (int i = 0; i < 2; ++i)
                    #pragma unroll
                    for (int j = 0; j < 5; ++j) S[i][j] = z;
            }
            #pragma unroll
            for (int kk = 0; kk < 2; ++kk) {
                int kc = kk * 32;
                v16h bk_[5];
                #pragma unroll
                for (int lt = 0; lt < 5; ++lt) bk_[lt] = loadB(khb + lt * 16 * HDIM, HDIM, kc, lane);
                #pragma unroll
                for (int mt = 0; mt < 2; ++mt) {
                    v16h a = loadA(qh + (th0 + mt * 16 + mrow) * XS + wav * 64, kc, lane);
                    #pragma unroll
                    for (int lt = 0; lt < 5; ++lt) S[mt][lt] = wmma16(a, bk_[lt], S[mt][lt]);
                }
            }
            // masked softmax over L=77 (rows in registers; 16-lane shuffle reductions per half)
            #pragma unroll
            for (int mt = 0; mt < 2; ++mt) {
                #pragma unroll
                for (int r = 0; r < 8; ++r) {
                    int tlq = mt * 16 + r + mhi;      // local row 0..31
                    float v[5]; float mx = -1e30f;
                    #pragma unroll
                    for (int j = 0; j < 5; ++j) {
                        int col = j * 16 + coln;
                        v[j] = (col < LTXT) ? S[mt][j][r] * scale : -1e30f;
                        mx = fmaxf(mx, v[j]);
                    }
                    #pragma unroll
                    for (int d = 1; d < 16; d <<= 1) mx = fmaxf(mx, __shfl_xor(mx, d, 32));
                    float e[5]; float sum = 0.f;
                    #pragma unroll
                    for (int j = 0; j < 5; ++j) {
                        int col = j * 16 + coln;
                        e[j] = (col < LTXT) ? __expf(v[j] - mx) : 0.f;
                        sum += e[j];
                    }
                    #pragma unroll
                    for (int d = 1; d < 16; d <<= 1) sum += __shfl_xor(sum, d, 32);
                    float inv = 1.0f / sum;
                    #pragma unroll
                    for (int j = 0; j < 5; ++j)
                        ps[wav][tlq * PSS + j * 16 + coln] = (_Float16)(e[j] * inv);
                }
            }
            // O = P * V  (K over LPAD=96)
            v8f O[2][4];
            {
                v8f z = {0.f,0.f,0.f,0.f,0.f,0.f,0.f,0.f};
                #pragma unroll
                for (int i = 0; i < 2; ++i)
                    #pragma unroll
                    for (int j = 0; j < 4; ++j) O[i][j] = z;
            }
            #pragma unroll
            for (int kk = 0; kk < 3; ++kk) {
                int kc = kk * 32;
                v16h bv_[4];
                #pragma unroll
                for (int nt = 0; nt < 4; ++nt) bv_[nt] = loadB(vb + nt * 16 * LPAD, LPAD, kc, lane);
                #pragma unroll
                for (int mt = 0; mt < 2; ++mt) {
                    v16h a = loadA(ps[wav] + (mt * 16 + mrow) * PSS, kc, lane);
                    #pragma unroll
                    for (int nt = 0; nt < 4; ++nt) O[mt][nt] = wmma16(a, bv_[nt], O[mt][nt]);
                }
            }
            // store attention output into xs_os[t][c] (reuse; GEMM1 reads ended at prior barrier)
            #pragma unroll
            for (int mt = 0; mt < 2; ++mt)
                #pragma unroll
                for (int nt = 0; nt < 4; ++nt)
                    #pragma unroll
                    for (int r = 0; r < 8; ++r) {
                        int tl = th0 + mt * 16 + r + mhi;
                        xs_os[tl * XS + wav * 64 + nt * 16 + (lane & 15)] = (_Float16)O[mt][nt][r];
                    }
        }
    }
    __syncthreads();

    // --- GEMM2: y[o][t] = wo * o_attn, + bias + residual, f32 out ---
    v8f Co[4][4];
    {
        v8f z = {0.f,0.f,0.f,0.f,0.f,0.f,0.f,0.f};
        #pragma unroll
        for (int i = 0; i < 4; ++i)
            #pragma unroll
            for (int j = 0; j < 4; ++j) Co[i][j] = z;
    }
    for (int kc = 0; kc < C_CH; kc += 32) {
        v16h bfr[4];
        #pragma unroll
        for (int tt = 0; tt < 4; ++tt) bfr[tt] = loadB(xs_os + tt * 16 * XS, XS, kc, lane);
        #pragma unroll
        for (int ot = 0; ot < 4; ++ot) {
            v16h a = loadA(wo16 + (size_t)(obase + ot * 16 + mrow) * C_CH, kc, lane);
            #pragma unroll
            for (int tt = 0; tt < 4; ++tt) Co[ot][tt] = wmma16(a, bfr[tt], Co[ot][tt]);
        }
    }
    {
        int n = lane & 15;
        #pragma unroll
        for (int tt = 0; tt < 4; ++tt) {
            int tg = t0 + tt * 16 + n;
            #pragma unroll
            for (int ot = 0; ot < 4; ++ot) {
                #pragma unroll
                for (int r = 0; r < 8; ++r) {
                    int o = obase + ot * 16 + r + mhi;
                    size_t idx = ((size_t)b * C_CH + o) * HWTOK + tg;
                    out[idx] = Co[ot][tt][r] + bo[o] + x[idx];
                }
            }
        }
    }
}

// ---------------------------------------------------------------- host launcher
extern "C" void kernel_launch(void* const* d_in, const int* in_sizes, int n_in,
                              void* d_out, int out_size, void* d_ws, size_t ws_size,
                              hipStream_t stream) {
    const float* x    = (const float*)d_in[0];
    const float* text = (const float*)d_in[1];
    const float* wq   = (const float*)d_in[2];
    const float* bq   = (const float*)d_in[3];
    const float* wk   = (const float*)d_in[4];
    const float* bk   = (const float*)d_in[5];
    const float* wv   = (const float*)d_in[6];
    const float* bv   = (const float*)d_in[7];
    const float* wo   = (const float*)d_in[8];
    const float* bo   = (const float*)d_in[9];
    const float* g1   = (const float*)d_in[10];
    const float* b1   = (const float*)d_in[11];
    const float* g2   = (const float*)d_in[12];
    const float* b2   = (const float*)d_in[13];
    float* out = (float*)d_out;

    char* ws = (char*)d_ws;
    _Float16* wq16 = (_Float16*)(ws + 0);          // 512KB
    _Float16* wo16 = (_Float16*)(ws + 524288);     // 512KB
    _Float16* wk16 = (_Float16*)(ws + 1048576);    // 768KB
    _Float16* wv16 = (_Float16*)(ws + 1835008);    // 768KB
    _Float16* kh   = (_Float16*)(ws + 2621440);    // 16*8*96*64*2 = 1.5MB
    _Float16* vhT  = (_Float16*)(ws + 4194304);    // 1.5MB
    float*    kf32 = (float*)   (ws + 5767168);    // 16*96*512*4 = 3MB

    cvt_f16_kernel<<<1024, 256, 0, stream>>>(wq, wq16, 512 * 512);
    cvt_f16_kernel<<<1024, 256, 0, stream>>>(wo, wo16, 512 * 512);
    cvt_f16_kernel<<<1536, 256, 0, stream>>>(wk, wk16, 512 * 768);
    cvt_f16_kernel<<<1536, 256, 0, stream>>>(wv, wv16, 512 * 768);
    kv_gemm_kernel<<<320, 256, 0, stream>>>(text, wk16, wv16, bk, bv, kf32, vhT);
    ln_k_kernel<<<16 * LPAD, 256, 0, stream>>>(kf32, g2, b2, kh, vhT);
    fused_attn_kernel<<<16 * 64, 256, 0, stream>>>(x, wq16, wo16, kh, vhT, bq, bo, g1, b1, out);
}
